// EdgePredictor_59631325937959
// MI455X (gfx1250) — matrix operand order repaired and checked by reading, not compiled, not run on hardware
//
#include <hip/hip_runtime.h>
#include <hip/hip_bf16.h>

typedef __attribute__((ext_vector_type(16))) _Float16 v16h;
typedef __attribute__((ext_vector_type(8)))  _Float16 v8h;
typedef __attribute__((ext_vector_type(4)))  _Float16 v4h;
typedef __attribute__((ext_vector_type(8)))  float    v8f;

#define DFEAT        128
#define TWO_D        256
#define LDS_STRIDE   272        // 256 + 16 f16 pad: 32B-aligned rows, no 16-way bank conflict
#define N_NODES      50000
#define N_EDGES_TOT  640000

// ---------------------------------------------------------------------------
// Pre-pass: convert x [50000x128] and W1 [128x256] fp32 -> f16 into workspace.
// ---------------------------------------------------------------------------
__global__ __launch_bounds__(256)
void cvt_to_f16_kernel(const float* __restrict__ x,
                       const float* __restrict__ W1,
                       _Float16* __restrict__ xh,
                       _Float16* __restrict__ w1h)
{
    const long NX4 = (long)N_NODES * DFEAT / 4;   // 1,600,000 float4s
    const long NW4 = (long)DFEAT * TWO_D / 4;     // 8,192 float4s
    long i = (long)blockIdx.x * 256 + threadIdx.x;
    if (i < NX4) {
        float4 f = ((const float4*)x)[i];
        v4h h; h[0]=(_Float16)f.x; h[1]=(_Float16)f.y; h[2]=(_Float16)f.z; h[3]=(_Float16)f.w;
        ((v4h*)xh)[i] = h;
    } else if (i < NX4 + NW4) {
        long j = i - NX4;
        float4 f = ((const float4*)W1)[j];
        v4h h; h[0]=(_Float16)f.x; h[1]=(_Float16)f.y; h[2]=(_Float16)f.z; h[3]=(_Float16)f.w;
        ((v4h*)w1h)[j] = h;
    }
}

// ---------------------------------------------------------------------------
// Main kernel: f16 gathers (32-bit GVS addressing), dual 16-edge M-tiles per
// wave (32 edges/wave), bias folded into WMMA accumulator init.
// ---------------------------------------------------------------------------
__global__ __launch_bounds__(256)
void edge_mlp_f16_kernel(const _Float16* __restrict__ xh,
                         const int*      __restrict__ src,
                         const int*      __restrict__ dst,
                         const _Float16* __restrict__ w1h,
                         const float*    __restrict__ b1,
                         const float*    __restrict__ W2,
                         const float*    __restrict__ b2,
                         float*          __restrict__ out)
{
    __shared__ alignas(32) _Float16 lW1[DFEAT * LDS_STRIDE];  // ~68 KB of 320 KB WGP LDS
    __shared__ float lb1[DFEAT];
    __shared__ float lw2d[DFEAT];

    const int tid = threadIdx.x;

    // Stage pre-converted W1 (f16) into padded LDS: 2048 chunks of 16 f16.
    for (int ci = tid; ci < (DFEAT * TWO_D) / 16; ci += 256) {
        int n = ci >> 4;
        int k = (ci & 15) * 16;
        *(v16h*)&lW1[n * LDS_STRIDE + k] = *(const v16h*)&w1h[n * TWO_D + k];
    }
    if (tid < DFEAT) {
        lb1[tid]  = b1[tid];
        lw2d[tid] = W2[DFEAT + tid] - W2[tid];   // softmax(2)[1] == sigmoid(z1-z0)
    }
    __syncthreads();

    const int wave = tid >> 5;
    const int lane = tid & 31;
    const int half = lane >> 4;   // A/B K-split half, C M-split half
    const int m    = lane & 15;

    const int e0 = (blockIdx.x * 8 + wave) * 32;   // 32 edges per wave
    const int eA = e0 + m;
    const int eB = e0 + 16 + m;

    // 32-bit byte offsets into xh (max 12.8MB): uniform SGPR base + VGPR u32
    // offset + folded immediates -> GVS-mode global_load_b128, no 64-bit VALU.
    const char* __restrict__ xb = (const char*)xh;
    const unsigned hskew = (unsigned)(half * 16);             // half's 8-f16 byte skew
    const unsigned osA = (((unsigned)src[eA]) << 8) + hskew;  // src row, tile A
    const unsigned odA = (((unsigned)dst[eA]) << 8) + hskew;  // dst row, tile A
    const unsigned osB = (((unsigned)src[eB]) << 8) + hskew;  // src row, tile B
    const unsigned odB = (((unsigned)dst[eB]) << 8) + hskew;  // dst row, tile B

    // A 16x32 f16 layout: lane(half=0) holds K {0..7,16..23}; half=1 holds {8..15,24..31}.
    v16h afA[8], afB[8];
#pragma unroll
    for (int c = 0; c < 8; ++c) {
        const unsigned co = (unsigned)((c & 3) * 64);          // chunk byte offset in row
        const unsigned oA = (c < 4) ? osA : odA;               // k<128 -> src, else dst
        const unsigned oB = (c < 4) ? osB : odB;
        v8h a_lo = *(const v8h*)(xb + (size_t)oA + co);
        v8h a_hi = *(const v8h*)(xb + (size_t)oA + co + 32);
        v8h b_lo = *(const v8h*)(xb + (size_t)oB + co);
        v8h b_hi = *(const v8h*)(xb + (size_t)oB + co + 32);
        afA[c] = __builtin_shufflevector(a_lo, a_hi, 0,1,2,3,4,5,6,7,8,9,10,11,12,13,14,15);
        afB[c] = __builtin_shufflevector(b_lo, b_hi, 0,1,2,3,4,5,6,7,8,9,10,11,12,13,14,15);
    }

    float pA[8], pB[8];
#pragma unroll
    for (int r = 0; r < 8; ++r) { pA[r] = 0.0f; pB[r] = 0.0f; }

#pragma unroll
    for (int nt = 0; nt < 8; ++nt) {
        const int n   = nt * 16 + m;        // this lane's output column
        const float bn = lb1[n];
        const float wn = lw2d[n];
        v8f accA, accB;
#pragma unroll
        for (int r = 0; r < 8; ++r) { accA[r] = bn; accB[r] = bn; }  // bias as C-init
#pragma unroll
        for (int c = 0; c < 8; ++c) {
            // B 32x16 f16: lane = column n, K local {0..7,16..23} (half=0) / {8..15,24..31} (half=1)
            const v8h bl8 = *(const v8h*)&lW1[n * LDS_STRIDE + c * 32 + half * 8];
            const v8h bh8 = *(const v8h*)&lW1[n * LDS_STRIDE + c * 32 + 16 + half * 8];
            const v16h bf = __builtin_shufflevector(bl8, bh8, 0,1,2,3,4,5,6,7,8,9,10,11,12,13,14,15);
            accA = __builtin_amdgcn_wmma_f32_16x16x32_f16(
                       false, afA[c], false, bf, (short)0, accA, false, false);
            accB = __builtin_amdgcn_wmma_f32_16x16x32_f16(
                       false, afB[c], false, bf, (short)0, accB, false, false);
        }
#pragma unroll
        for (int r = 0; r < 8; ++r) {       // C layout: lane=N, VGPR r -> M = r + 8*half
            float hA = accA[r] > 0.0f ? accA[r] : 0.0f;
            float hB = accB[r] > 0.0f ? accB[r] : 0.0f;
            pA[r] = fmaf(hA, wn, pA[r]);
            pB[r] = fmaf(hB, wn, pB[r]);
        }
    }

    // Reduce across the 16 lanes sharing a row (xor masks stay within each half).
    const float b2d = b2[1] - b2[0];
#pragma unroll
    for (int r = 0; r < 8; ++r) {
        float vA = pA[r], vB = pB[r];
        vA += __shfl_xor(vA, 8, 32);  vB += __shfl_xor(vB, 8, 32);
        vA += __shfl_xor(vA, 4, 32);  vB += __shfl_xor(vB, 4, 32);
        vA += __shfl_xor(vA, 2, 32);  vB += __shfl_xor(vB, 2, 32);
        vA += __shfl_xor(vA, 1, 32);  vB += __shfl_xor(vB, 1, 32);
        if (m == 0) {
            const unsigned row = (unsigned)(e0 + half * 8 + r);
            out[row]      = 1.0f / (1.0f + __expf(-(vA + b2d)));
            out[row + 16] = 1.0f / (1.0f + __expf(-(vB + b2d)));
        }
    }
}

// ---------------------------------------------------------------------------
// Fallback (no workspace): self-contained fp32-input version.
// ---------------------------------------------------------------------------
__global__ __launch_bounds__(256)
void edge_mlp_f32in_kernel(const float* __restrict__ x,
                           const int*   __restrict__ src,
                           const int*   __restrict__ dst,
                           const float* __restrict__ W1,
                           const float* __restrict__ b1,
                           const float* __restrict__ W2,
                           const float* __restrict__ b2,
                           float*       __restrict__ out)
{
    __shared__ alignas(32) _Float16 lW1[DFEAT * LDS_STRIDE];
    __shared__ float lb1[DFEAT];
    __shared__ float lw2d[DFEAT];

    const int tid = threadIdx.x;
    const float2* W1v = (const float2*)W1;
    for (int i2 = tid; i2 < (DFEAT * TWO_D) / 2; i2 += 256) {
        float2 f = W1v[i2];
        int n = i2 >> 7, k = (i2 & 127) * 2;
        lW1[n * LDS_STRIDE + k + 0] = (_Float16)f.x;
        lW1[n * LDS_STRIDE + k + 1] = (_Float16)f.y;
    }
    if (tid < DFEAT) { lb1[tid] = b1[tid]; lw2d[tid] = W2[DFEAT + tid] - W2[tid]; }
    __syncthreads();

    const int wave = tid >> 5, lane = tid & 31, half = lane >> 4, m = lane & 15;
    const int e0 = (blockIdx.x * 8 + wave) * 16;
    const int e  = e0 + m;
    const float* xs = x + (long)src[e] * DFEAT;
    const float* xd = x + (long)dst[e] * DFEAT;

    v16h afrag[8];
#pragma unroll
    for (int c = 0; c < 8; ++c) {
        const float* row = (c < 4) ? xs : xd;
        const int kb = (c & 3) * 32 + half * 8;
        float4 f0 = *(const float4*)(row + kb);
        float4 f1 = *(const float4*)(row + kb + 4);
        float4 f2 = *(const float4*)(row + kb + 16);
        float4 f3 = *(const float4*)(row + kb + 20);
        v16h a;
        a[0]=(_Float16)f0.x; a[1]=(_Float16)f0.y; a[2]=(_Float16)f0.z; a[3]=(_Float16)f0.w;
        a[4]=(_Float16)f1.x; a[5]=(_Float16)f1.y; a[6]=(_Float16)f1.z; a[7]=(_Float16)f1.w;
        a[8]=(_Float16)f2.x; a[9]=(_Float16)f2.y; a[10]=(_Float16)f2.z; a[11]=(_Float16)f2.w;
        a[12]=(_Float16)f3.x; a[13]=(_Float16)f3.y; a[14]=(_Float16)f3.z; a[15]=(_Float16)f3.w;
        afrag[c] = a;
    }

    float partial[8];
#pragma unroll
    for (int r = 0; r < 8; ++r) partial[r] = 0.0f;
#pragma unroll
    for (int nt = 0; nt < 8; ++nt) {
        const int n = nt * 16 + m;
        const float bn = lb1[n], wn = lw2d[n];
        v8f acc;
#pragma unroll
        for (int r = 0; r < 8; ++r) acc[r] = bn;
#pragma unroll
        for (int c = 0; c < 8; ++c) {
            const v8h bl8 = *(const v8h*)&lW1[n * LDS_STRIDE + c * 32 + half * 8];
            const v8h bh8 = *(const v8h*)&lW1[n * LDS_STRIDE + c * 32 + 16 + half * 8];
            const v16h bf = __builtin_shufflevector(bl8, bh8, 0,1,2,3,4,5,6,7,8,9,10,11,12,13,14,15);
            acc = __builtin_amdgcn_wmma_f32_16x16x32_f16(
                      false, afrag[c], false, bf, (short)0, acc, false, false);
        }
#pragma unroll
        for (int r = 0; r < 8; ++r) {
            float h = acc[r] > 0.0f ? acc[r] : 0.0f;
            partial[r] = fmaf(h, wn, partial[r]);
        }
    }
    const float b2d = b2[1] - b2[0];
#pragma unroll
    for (int r = 0; r < 8; ++r) {
        float v = partial[r];
        v += __shfl_xor(v, 8, 32); v += __shfl_xor(v, 4, 32);
        v += __shfl_xor(v, 2, 32); v += __shfl_xor(v, 1, 32);
        if (m == 0)
            out[e0 + half * 8 + r] = 1.0f / (1.0f + __expf(-(v + b2d)));
    }
}

extern "C" void kernel_launch(void* const* d_in, const int* in_sizes, int n_in,
                              void* d_out, int out_size, void* d_ws, size_t ws_size,
                              hipStream_t stream) {
    const float* x   = (const float*)d_in[0];
    const int*   src = (const int*)  d_in[1];
    const int*   dst = (const int*)  d_in[2];
    const float* W1  = (const float*)d_in[3];
    const float* b1  = (const float*)d_in[4];
    const float* W2  = (const float*)d_in[5];
    const float* b2  = (const float*)d_in[6];
    float* out = (float*)d_out;

    const size_t xh_bytes  = (size_t)N_NODES * DFEAT * sizeof(_Float16);   // 12.8 MB
    const size_t w1h_bytes = (size_t)DFEAT * TWO_D * sizeof(_Float16);     // 64 KB

    if (ws_size >= xh_bytes + w1h_bytes) {
        _Float16* xh  = (_Float16*)d_ws;
        _Float16* w1h = (_Float16*)((char*)d_ws + xh_bytes);
        const long total4 = (long)N_NODES * DFEAT / 4 + (long)DFEAT * TWO_D / 4;
        const int cvt_blocks = (int)((total4 + 255) / 256);
        cvt_to_f16_kernel<<<cvt_blocks, 256, 0, stream>>>(x, W1, xh, w1h);
        edge_mlp_f16_kernel<<<N_EDGES_TOT / 256, 256, 0, stream>>>(
            xh, src, dst, w1h, b1, W2, b2, out);
    } else {
        edge_mlp_f32in_kernel<<<N_EDGES_TOT / 128, 256, 0, stream>>>(
            x, src, dst, W1, b1, W2, b2, out);
    }
}
